// NodeModel_7060926234899
// MI455X (gfx1250) — matrix operand order repaired and checked by reading, not compiled, not run on hardware
//
#include <hip/hip_runtime.h>

typedef __attribute__((ext_vector_type(16))) _Float16 v16h;
typedef __attribute__((ext_vector_type(8)))  _Float16 v8h;
typedef __attribute__((ext_vector_type(8)))  float    v8f;

// Packed-weight layout (halves) inside d_ws:
//  pW1a: frags 0..7    (K=64)   offset 0
//  pW1b: frags 0..7    (K=64)   offset 4096
//  pW2a: frags 0..11   (K=96)   offset 8192
//  pW2b: frags 0..7    (K=64)   offset 14336
// Each fragment: 512 halves (1 KB), lane-major: lane L owns halves [16L,16L+16).
#define PACK_W1A 0
#define PACK_W1B 4096
#define PACK_W2A 8192
#define PACK_W2B 14336
#define PACK_HALVES 18432
#define WS_DATA_OFFSET 65536   // bytes; summed/cnt live after packed weights

__device__ __forceinline__ v8f wmma_f16(v16h a, v16h b, v8f c) {
  return __builtin_amdgcn_wmma_f32_16x16x32_f16(
      false, a, false, b, (short)0, c, false, false);
}

__device__ __forceinline__ v16h cat8(v8h lo, v8h hi) {
  return __builtin_shufflevector(lo, hi, 0, 1, 2, 3, 4, 5, 6, 7,
                                 8, 9, 10, 11, 12, 13, 14, 15);
}

// A-matrix 16x32 f16 fragment from row-major LDS [rows][stride] halves.
// ISA 7.12.2: lane = 16*hi + m; halves j=0..7 -> k = 8*hi + j,
// j=8..15 -> k = 16 + 8*hi + (j-8). Two contiguous 16B loads.
__device__ __forceinline__ v16h load_A_frag(const _Float16* s, int stride,
                                            int m0, int k0, int lane) {
  const _Float16* row =
      s + (m0 + (lane & 15)) * stride + k0 + ((lane >> 4) << 3);
  v8h lo = *(const v8h*)row;
  v8h hi = *(const v8h*)(row + 16);
  return cat8(lo, hi);
}

// B-matrix 32x16 f16 fragment from packed global weights: 2x global_load_b128.
__device__ __forceinline__ v16h load_B_packed(const _Float16* base, int frag,
                                              int lane) {
  const _Float16* p = base + frag * 512 + lane * 16;
  v8h lo = *(const v8h*)p;
  v8h hi = *(const v8h*)(p + 8);
  return cat8(lo, hi);
}

// C/D f32 16x16: VGPR r holds M = 8*(lane>>4) + r, N = lane&15.
__device__ __forceinline__ v8f bias_init(const float* b, int n0, int lane) {
  float bv = b[n0 + (lane & 15)];
  v8f c;
#pragma unroll
  for (int r = 0; r < 8; ++r) c[r] = bv;
  return c;
}

__device__ __forceinline__ void store_D_half(v8f c, _Float16* s, int stride,
                                             int m0, int n0, int lane,
                                             bool relu) {
  int n = n0 + (lane & 15);
  int mb = m0 + ((lane >> 4) << 3);
#pragma unroll
  for (int r = 0; r < 8; ++r) {
    float v = c[r];
    if (relu) v = fmaxf(v, 0.0f);
    s[(mb + r) * stride + n] = (_Float16)v;
  }
}

__device__ __forceinline__ v8h cvt8(float4 a, float4 b) {
  v8h h;
  h[0] = (_Float16)a.x; h[1] = (_Float16)a.y;
  h[2] = (_Float16)a.z; h[3] = (_Float16)a.w;
  h[4] = (_Float16)b.x; h[5] = (_Float16)b.y;
  h[6] = (_Float16)b.z; h[7] = (_Float16)b.w;
  return h;
}

// ---------------------------------------------------------------------------
// Weight packer: f32 row-major [K][64] -> f16 WMMA B-fragment order.
// frag f = ks*4 + nt; element (lane, j): k = ks*32 + 16*(j>>3) + 8*(lane>>4) + (j&7),
// n = nt*16 + (lane&15).
// ---------------------------------------------------------------------------
__device__ __forceinline__ void pack_one(const float* __restrict__ W, int K,
                                         _Float16* __restrict__ dst, int t) {
  int total = K * 64;
  for (int idx = t; idx < total; idx += 256) {
    int f = idx >> 9;
    int r = idx & 511;
    int lane = r >> 4;
    int j = r & 15;
    int ks = f >> 2, nt = f & 3;
    int k = ks * 32 + ((j >> 3) << 4) + ((lane >> 4) << 3) + (j & 7);
    int n = nt * 16 + (lane & 15);
    dst[idx] = (_Float16)W[k * 64 + n];
  }
}

__global__ __launch_bounds__(256) void pack_weights(
    const float* __restrict__ W1a, const float* __restrict__ W1b,
    const float* __restrict__ W2a, const float* __restrict__ W2b,
    _Float16* __restrict__ out) {
  int t = threadIdx.x;
  pack_one(W1a, 64, out + PACK_W1A, t);
  pack_one(W1b, 64, out + PACK_W1B, t);
  pack_one(W2a, 96, out + PACK_W2A, t);
  pack_one(W2b, 64, out + PACK_W2B, t);
}

// ---------------------------------------------------------------------------
// Edge kernel: gather msg = x[row] || edge_attr, h = relu(msg@W1a+b1a)@W1b+b1b,
// atomic scatter into summed[col], count into cnt.
// ---------------------------------------------------------------------------
__global__ __launch_bounds__(256) void edge_mlp_scatter(
    const float* __restrict__ x, const int* __restrict__ ei,
    const float* __restrict__ ea, const _Float16* __restrict__ wpack,
    const float* __restrict__ b1a, const float* __restrict__ b1b,
    float* __restrict__ summed, float* __restrict__ cnt, int n_edges) {
  __shared__ _Float16 sMsg[128 * 64];  // reused as h1 after layer 1
  __shared__ int sCol[128];

  const int t = threadIdx.x;
  const int e0 = blockIdx.x * 128;

  // Stage msg (f32 -> f16), vectorized: 2 threads per edge.
  {
    int e = t >> 1, part = t & 1;
    int eg = e0 + e;
    v8h* dst = (v8h*)&sMsg[e * 64 + part * 32];
    if (eg < n_edges) {
      const float4* src;
      if (part == 0) {
        int r = ei[eg];
        src = (const float4*)(x + (size_t)r * 32);
        int c = ei[n_edges + eg];
        sCol[e] = c;
        unsafeAtomicAdd(&cnt[c], 1.0f);
      } else {
        src = (const float4*)(ea + (size_t)eg * 32);
      }
#pragma unroll
      for (int q = 0; q < 4; ++q) dst[q] = cvt8(src[2 * q], src[2 * q + 1]);
    } else {
      v8h z = (v8h)(_Float16)0.0f;
#pragma unroll
      for (int q = 0; q < 4; ++q) dst[q] = z;
    }
  }
  __syncthreads();

  const int lane = t & 31;
  const int m0 = (t >> 5) * 16;
  const _Float16* pW1a = wpack + PACK_W1A;
  const _Float16* pW1b = wpack + PACK_W1B;

  // Layer 1: h1 = relu(msg @ W1a + b1a)
  v16h a0 = load_A_frag(sMsg, 64, m0, 0, lane);
  v16h a1 = load_A_frag(sMsg, 64, m0, 32, lane);
  v8f acc[4];
#pragma unroll
  for (int nt = 0; nt < 4; ++nt) {
    v8f c = bias_init(b1a, nt * 16, lane);
    c = wmma_f16(a0, load_B_packed(pW1a, nt, lane), c);
    c = wmma_f16(a1, load_B_packed(pW1a, 4 + nt, lane), c);
    acc[nt] = c;
  }
#pragma unroll
  for (int nt = 0; nt < 4; ++nt)
    store_D_half(acc[nt], sMsg, 64, m0, nt * 16, lane, true);
  __syncthreads();

  // Layer 2: h2 = h1 @ W1b + b1b
  a0 = load_A_frag(sMsg, 64, m0, 0, lane);
  a1 = load_A_frag(sMsg, 64, m0, 32, lane);
#pragma unroll
  for (int nt = 0; nt < 4; ++nt) {
    v8f c = bias_init(b1b, nt * 16, lane);
    c = wmma_f16(a0, load_B_packed(pW1b, nt, lane), c);
    c = wmma_f16(a1, load_B_packed(pW1b, 4 + nt, lane), c);
    acc[nt] = c;
  }

  // Atomic scatter h2 -> summed[col].
  const int mb = (lane >> 4) << 3;
#pragma unroll
  for (int nt = 0; nt < 4; ++nt) {
    int n = nt * 16 + (lane & 15);
#pragma unroll
    for (int r = 0; r < 8; ++r) {
      int e = m0 + mb + r;
      if (e0 + e < n_edges) {
        unsafeAtomicAdd(&summed[(size_t)sCol[e] * 64 + n], acc[nt][r]);
      }
    }
  }
}

// ---------------------------------------------------------------------------
// Node kernel: z = x || summed/max(cnt,1); out = relu(z@W2a+b2a)@W2b+b2b.
// ---------------------------------------------------------------------------
__global__ __launch_bounds__(256) void node_mlp(
    const float* __restrict__ x, const float* __restrict__ summed,
    const float* __restrict__ cnt, const _Float16* __restrict__ wpack,
    const float* __restrict__ b2a, const float* __restrict__ b2b,
    float* __restrict__ out, int n_nodes) {
  __shared__ _Float16 sZ[128 * 96];  // cols 0..63 reused for h1

  const int t = threadIdx.x;
  const int node0 = blockIdx.x * 128;

  // Stage z: 2 threads per node (part0: x[0..31], part1: agg[0..63]).
  {
    int i = t >> 1, part = t & 1;
    int node = node0 + i;
    if (node < n_nodes) {
      if (part == 0) {
        v8h* dst = (v8h*)&sZ[i * 96];
        const float4* src = (const float4*)(x + (size_t)node * 32);
#pragma unroll
        for (int q = 0; q < 4; ++q) dst[q] = cvt8(src[2 * q], src[2 * q + 1]);
      } else {
        v8h* dst = (v8h*)&sZ[i * 96 + 32];
        float inv = 1.0f / fmaxf(cnt[node], 1.0f);
        const float4* src = (const float4*)(summed + (size_t)node * 64);
#pragma unroll
        for (int q = 0; q < 8; ++q) {
          float4 a = src[2 * q], b = src[2 * q + 1];
          a.x *= inv; a.y *= inv; a.z *= inv; a.w *= inv;
          b.x *= inv; b.y *= inv; b.z *= inv; b.w *= inv;
          dst[q] = cvt8(a, b);
        }
      }
    } else {
      v8h* dst = (v8h*)&sZ[i * 96 + part * 48];
      v8h z = (v8h)(_Float16)0.0f;
#pragma unroll
      for (int q = 0; q < 6; ++q) dst[q] = z;
    }
  }
  __syncthreads();

  const int lane = t & 31;
  const int m0 = (t >> 5) * 16;
  const _Float16* pW2a = wpack + PACK_W2A;
  const _Float16* pW2b = wpack + PACK_W2B;

  // Layer A: h = relu(z @ W2a + b2a), K = 96.
  v16h a0 = load_A_frag(sZ, 96, m0, 0, lane);
  v16h a1 = load_A_frag(sZ, 96, m0, 32, lane);
  v16h a2 = load_A_frag(sZ, 96, m0, 64, lane);
  v8f acc[4];
#pragma unroll
  for (int nt = 0; nt < 4; ++nt) {
    v8f c = bias_init(b2a, nt * 16, lane);
    c = wmma_f16(a0, load_B_packed(pW2a, nt, lane), c);
    c = wmma_f16(a1, load_B_packed(pW2a, 4 + nt, lane), c);
    c = wmma_f16(a2, load_B_packed(pW2a, 8 + nt, lane), c);
    acc[nt] = c;
  }
#pragma unroll
  for (int nt = 0; nt < 4; ++nt)
    store_D_half(acc[nt], sZ, 96, m0, nt * 16, lane, true);
  __syncthreads();

  // Layer B: out = h @ W2b + b2b.
  a0 = load_A_frag(sZ, 96, m0, 0, lane);
  a1 = load_A_frag(sZ, 96, m0, 32, lane);
  const int mb = (lane >> 4) << 3;
#pragma unroll
  for (int nt = 0; nt < 4; ++nt) {
    v8f c = bias_init(b2b, nt * 16, lane);
    c = wmma_f16(a0, load_B_packed(pW2b, nt, lane), c);
    c = wmma_f16(a1, load_B_packed(pW2b, 4 + nt, lane), c);
    int n = nt * 16 + (lane & 15);
#pragma unroll
    for (int r = 0; r < 8; ++r) {
      int node = node0 + m0 + mb + r;
      if (node < n_nodes) out[(size_t)node * 64 + n] = c[r];
    }
  }
}

// ---------------------------------------------------------------------------
extern "C" void kernel_launch(void* const* d_in, const int* in_sizes, int n_in,
                              void* d_out, int out_size, void* d_ws,
                              size_t ws_size, hipStream_t stream) {
  const float* x = (const float*)d_in[0];
  const int* ei = (const int*)d_in[1];
  const float* ea = (const float*)d_in[2];
  const float* W1a = (const float*)d_in[5];
  const float* b1a = (const float*)d_in[6];
  const float* W1b = (const float*)d_in[7];
  const float* b1b = (const float*)d_in[8];
  const float* W2a = (const float*)d_in[9];
  const float* b2a = (const float*)d_in[10];
  const float* W2b = (const float*)d_in[11];
  const float* b2b = (const float*)d_in[12];

  const int n_nodes = in_sizes[0] / 32;  // x: [N, 32]
  const int n_edges = in_sizes[1] / 2;   // edge_index: [2, E]

  _Float16* wpack = (_Float16*)d_ws;
  float* summed = (float*)((char*)d_ws + WS_DATA_OFFSET);  // [N, 64]
  float* cnt = summed + (size_t)n_nodes * 64;              // [N]

  size_t zero_bytes = ((size_t)n_nodes * 64 + (size_t)n_nodes) * sizeof(float);
  hipMemsetAsync((char*)d_ws + WS_DATA_OFFSET, 0, zero_bytes, stream);

  pack_weights<<<1, 256, 0, stream>>>(W1a, W1b, W2a, W2b, wpack);

  int eblocks = (n_edges + 127) / 128;
  edge_mlp_scatter<<<eblocks, 256, 0, stream>>>(x, ei, ea, wpack, b1a, b1b,
                                                summed, cnt, n_edges);

  int nblocks = (n_nodes + 127) / 128;
  node_mlp<<<nblocks, 256, 0, stream>>>(x, summed, cnt, wpack, b2a, b2b,
                                        (float*)d_out, n_nodes);
}
